// MultiHeadAttention_15496242004786
// MI455X (gfx1250) — matrix operand order repaired and checked by reading, not compiled, not run on hardware
//
#include <hip/hip_runtime.h>

// ---------------------------------------------------------------------------
// MHA forward for gfx1250 (MI455X): bf16 WMMA (V_WMMA_F32_16X16X32_BF16),
// flash-style attention, wave32 layouts per CDNA5 ISA 7.12.2.
// All f32->bf16 conversion + transposition hoisted into one pre-pass so the
// GEMM main loops are pure b128 copy -> ds_load_b128 -> wmma.
// ---------------------------------------------------------------------------

typedef __attribute__((ext_vector_type(16))) __bf16 v16bf;
typedef __attribute__((ext_vector_type(8)))  float  v8f;

constexpr int C_  = 1024;   // embed dim
constexpr int T_  = 2048;   // sequence length
constexpr int H_  = 16;     // heads
constexpr int D_  = 64;     // head dim
constexpr int N3_ = 3 * C_; // qkv out dim
constexpr int B_  = 4;

__device__ __forceinline__ v8f wmma_bf16(v16bf a, v16bf b, v8f c) {
  return __builtin_amdgcn_wmma_f32_16x16x32_bf16(false, a, false, b,
                                                 (short)0, c, false, false);
}

union U16x16 { v16bf v; unsigned u[8]; };

// A fragment (16x32 bf16, M x K) from LDS laid out As[m][k], row stride lda.
__device__ __forceinline__ v16bf load_a_frag(const __bf16* Asrc, int lda, int lane) {
  const int hl = lane >> 4, m = lane & 15;
  const unsigned* p = (const unsigned*)(Asrc + m * lda + hl * 8);
  U16x16 r;
#pragma unroll
  for (int i = 0; i < 4; ++i) { r.u[i] = p[i]; r.u[4 + i] = p[8 + i]; }
  return r.v;
}

// B fragment (32x16 bf16, K x N) from LDS laid out transposed Bt[n][k].
__device__ __forceinline__ v16bf load_b_frag(const __bf16* Bsrc, int ldk, int lane) {
  const int hl = lane >> 4, n = lane & 15;
  const unsigned* p = (const unsigned*)(Bsrc + n * ldk + hl * 8);
  U16x16 r;
#pragma unroll
  for (int i = 0; i < 4; ++i) { r.u[i] = p[i]; r.u[4 + i] = p[8 + i]; }
  return r.v;
}

// ---------------------------------------------------------------------------
// Kernel 0: transpose + convert  src[rows][cols] f32  ->  dst[cols][rows] bf16
// 32x32 tiles, coalesced f32 reads along cols, coalesced bf16 writes.
// ---------------------------------------------------------------------------
__global__ __launch_bounds__(256) void tcvt_kernel(
    const float* __restrict__ src, __bf16* __restrict__ dst,
    int rows, int cols, size_t sBatch, size_t dBatch) {
  __shared__ float tile[32][33];
  src += (size_t)blockIdx.z * sBatch;
  dst += (size_t)blockIdx.z * dBatch;
  const int tid = threadIdx.x, tx = tid & 31, ty = tid >> 5;
  const int r0 = blockIdx.y * 32, c0 = blockIdx.x * 32;
#pragma unroll
  for (int i = 0; i < 4; ++i) {
    int r = ty + i * 8;
    tile[r][tx] = src[(size_t)(r0 + r) * cols + c0 + tx];
  }
  __syncthreads();
#pragma unroll
  for (int i = 0; i < 4; ++i) {
    int r = ty + i * 8;
    dst[(size_t)(c0 + r) * rows + r0 + tx] = (__bf16)tile[tx][r];
  }
}

// ---------------------------------------------------------------------------
// Kernel 1: QKV projection. qkv[b,t,n] = sum_c xbf[b,t,c]*wqkvT[n,c] + bias[n]
// A=xbf and B=wqkvT are both contiguous along K -> pure b128 staging.
// Block 256 (8 waves), tile 128(M=t) x 128(N=n), K chunks of 64 (2 wmma k-steps).
// ---------------------------------------------------------------------------
__global__ __launch_bounds__(256) void qkv_gemm_kernel(
    const __bf16* __restrict__ xbf, const __bf16* __restrict__ wqkvT,
    const float* __restrict__ bias,
    __bf16* __restrict__ Qb, __bf16* __restrict__ Kb, __bf16* __restrict__ Vb) {
  __shared__ __bf16 As[128 * 72];  // As[m][k]
  __shared__ __bf16 Bs[128 * 72];  // Bs[n][k]
  const int tid = threadIdx.x, lane = tid & 31, wv = tid >> 5;
  const int wm = wv >> 1, wn = wv & 1;
  const int b = blockIdx.z, m0 = blockIdx.y * 128, n0 = blockIdx.x * 128;
  v8f acc[2][4] = {};
  const __bf16* xrow = xbf + ((size_t)b * T_ + m0) * C_;
  const __bf16* wrow = wqkvT + (size_t)n0 * C_;
#pragma unroll 1
  for (int kt = 0; kt < C_ / 64; ++kt) {
    const __bf16* xs = xrow + kt * 64;
    const __bf16* ws = wrow + kt * 64;
    __builtin_prefetch(xs + 64, 0, 1);
    __builtin_prefetch(ws + 64, 0, 1);
#pragma unroll
    for (int i = 0; i < 4; ++i) {       // 128 rows x 8 uint4 each, x2 matrices
      int idx = tid + i * 256;
      int m = idx >> 3, kv = (idx & 7) * 8;
      *(uint4*)(As + m * 72 + kv) = *(const uint4*)(xs + (size_t)m * C_ + kv);
      *(uint4*)(Bs + m * 72 + kv) = *(const uint4*)(ws + (size_t)m * C_ + kv);
    }
    __syncthreads();
#pragma unroll
    for (int kc = 0; kc < 2; ++kc) {
      v16bf a0 = load_a_frag(As + (wm * 32 + 0) * 72 + kc * 32, 72, lane);
      v16bf a1 = load_a_frag(As + (wm * 32 + 16) * 72 + kc * 32, 72, lane);
#pragma unroll
      for (int j = 0; j < 4; ++j) {
        v16bf bf = load_b_frag(Bs + (wn * 64 + j * 16) * 72 + kc * 32, 72, lane);
        acc[0][j] = wmma_bf16(a0, bf, acc[0][j]);
        acc[1][j] = wmma_bf16(a1, bf, acc[1][j]);
      }
    }
    __syncthreads();
  }
  // Epilogue: bias + scatter to Q/K/V [B,H,T,D] bf16.
  const int hl = lane >> 4, nn = lane & 15;
#pragma unroll
  for (int j = 0; j < 4; ++j) {
    int n = n0 + wn * 64 + j * 16 + nn;
    float bv = bias[n];
    int sel = n >> 10, cp = n & (C_ - 1), h = cp >> 6, d = cp & (D_ - 1);
    __bf16* dst = (sel == 0) ? Qb : (sel == 1) ? Kb : Vb;
    size_t base = (((size_t)b * H_ + h) * T_) * D_ + d;
#pragma unroll
    for (int i = 0; i < 2; ++i)
#pragma unroll
      for (int r = 0; r < 8; ++r) {
        int t = m0 + wm * 32 + i * 16 + r + hl * 8;
        dst[base + (size_t)t * D_] = (__bf16)(acc[i][j][r] + bv);
      }
  }
}

// ---------------------------------------------------------------------------
// Kernel 2: flash attention. One block per (b, h, 128-query tile); 8 waves,
// each wave owns 16 query rows. Key tiles of 64; online softmax in registers
// (a score row spans the 16 lanes of a wave half -> shfl_xor reductions).
// ---------------------------------------------------------------------------
__global__ __launch_bounds__(256) void attn_kernel(
    const __bf16* __restrict__ Qb, const __bf16* __restrict__ Kb,
    const __bf16* __restrict__ Vb, __bf16* __restrict__ Ob) {
  __shared__ __bf16 Qs[128 * 72];      // Qs[q][d], pre-scaled by 1/sqrt(D)
  __shared__ __bf16 Ks[64 * 72];       // Ks[s][d]  (B^T layout for Q.K^T)
  __shared__ __bf16 Vt[64 * 72];       // Vt[d][s]  (B^T layout for P.V)
  __shared__ __bf16 Pb[8 * 16 * 72];   // per-wave P rows [16][72]
  union BF2 { unsigned u; __bf16 h[2]; };
  const int tid = threadIdx.x, lane = tid & 31, wv = tid >> 5;
  const int hl = lane >> 4, nn = lane & 15;
  const int b = blockIdx.z, h = blockIdx.y, q0 = blockIdx.x * 128;
  const size_t bh = (size_t)b * H_ + h;
  // Stage Q tile (scale by 0.125 = 1/sqrt(64) while copying).
#pragma unroll
  for (int i = 0; i < 8; ++i) {
    int idx = tid + i * 256;            // 128 rows x 16 dwords
    int qr = idx >> 4, k2 = (idx & 15) * 2;
    BF2 u; u.u = *(const unsigned*)(Qb + (bh * T_ + q0 + qr) * D_ + k2);
    BF2 o2;
    o2.h[0] = (__bf16)((float)u.h[0] * 0.125f);
    o2.h[1] = (__bf16)((float)u.h[1] * 0.125f);
    *(unsigned*)(Qs + qr * 72 + k2) = o2.u;
  }
  float mrow[8], lrow[8];
#pragma unroll
  for (int r = 0; r < 8; ++r) { mrow[r] = -3.0e38f; lrow[r] = 0.0f; }
  v8f o[4] = {};
  __syncthreads();
  // Q fragments are loop-invariant: hoist out of the key loop.
  const v16bf aq0 = load_a_frag(Qs + (wv * 16) * 72 + 0,  72, lane);
  const v16bf aq1 = load_a_frag(Qs + (wv * 16) * 72 + 32, 72, lane);
#pragma unroll 1
  for (int s0 = 0; s0 < T_; s0 += 64) {
    __syncthreads();  // previous tile's P.V reads of Vt are done
#pragma unroll
    for (int i = 0; i < 4; ++i) {
      int idx = tid + i * 256;          // 64 rows x 16 dwords
      int sr = idx >> 4, k2 = (idx & 15) * 2;
      *(unsigned*)(Ks + sr * 72 + k2) =
          *(const unsigned*)(Kb + (bh * T_ + s0 + sr) * D_ + k2);
      BF2 vv; vv.u = *(const unsigned*)(Vb + (bh * T_ + s0 + sr) * D_ + k2);
      Vt[(k2 + 0) * 72 + sr] = vv.h[0];
      Vt[(k2 + 1) * 72 + sr] = vv.h[1];
    }
    __syncthreads();
    // S = Q * K^T  (16 q-rows x 64 keys per wave)
    v8f sc[4] = {};
#pragma unroll
    for (int j = 0; j < 4; ++j) {
      v16bf b0 = load_b_frag(Ks + (j * 16) * 72 + 0,  72, lane);
      v16bf b1 = load_b_frag(Ks + (j * 16) * 72 + 32, 72, lane);
      sc[j] = wmma_bf16(aq0, b0, sc[j]);
      sc[j] = wmma_bf16(aq1, b1, sc[j]);
    }
    // Online softmax: row (M=r + 8*hl) spans 16 lanes of this half.
    float fac[8];
#pragma unroll
    for (int r = 0; r < 8; ++r) {
      float m = fmaxf(fmaxf(sc[0][r], sc[1][r]), fmaxf(sc[2][r], sc[3][r]));
#pragma unroll
      for (int d = 1; d < 16; d <<= 1) m = fmaxf(m, __shfl_xor(m, d));
      float mn = fmaxf(mrow[r], m);
      fac[r] = __expf(mrow[r] - mn);
      mrow[r] = mn;
      float sum = 0.0f;
#pragma unroll
      for (int j = 0; j < 4; ++j) {
        float p = __expf(sc[j][r] - mn);
        sum += p;
        Pb[(wv * 16 + r + hl * 8) * 72 + j * 16 + nn] = (__bf16)p;
      }
#pragma unroll
      for (int d = 1; d < 16; d <<= 1) sum += __shfl_xor(sum, d);
      lrow[r] = lrow[r] * fac[r] + sum;
    }
#pragma unroll
    for (int j = 0; j < 4; ++j)
#pragma unroll
      for (int r = 0; r < 8; ++r) o[j][r] *= fac[r];
    asm volatile("" ::: "memory");  // keep Pb stores before Pb fragment loads
    // O += P * V
#pragma unroll
    for (int kc = 0; kc < 2; ++kc) {
      v16bf ap = load_a_frag(Pb + (wv * 16) * 72 + kc * 32, 72, lane);
#pragma unroll
      for (int j = 0; j < 4; ++j) {
        v16bf bv = load_b_frag(Vt + (j * 16) * 72 + kc * 32, 72, lane);
        o[j] = wmma_bf16(ap, bv, o[j]);
      }
    }
  }
  // Normalize and store O [B,T,C] bf16.
#pragma unroll
  for (int r = 0; r < 8; ++r) lrow[r] = 1.0f / lrow[r];
#pragma unroll
  for (int j = 0; j < 4; ++j)
#pragma unroll
    for (int r = 0; r < 8; ++r) {
      int t = q0 + wv * 16 + r + hl * 8;
      Ob[((size_t)b * T_ + t) * C_ + h * D_ + j * 16 + nn] =
          (__bf16)(o[j][r] * lrow[r]);
    }
}

// ---------------------------------------------------------------------------
// Kernel 3: output projection  out[b,c,t] = sum_k O[b,t,k]*woutT[c,k] + bias[c]
// Two-pass LDS transpose (union'd with A/B tiles) -> coalesced f32 stores
// along t in the [B,C,T] output.
// ---------------------------------------------------------------------------
__global__ __launch_bounds__(256) void oproj_kernel(
    const __bf16* __restrict__ Ob, const __bf16* __restrict__ woutT,
    const float* __restrict__ bias, float* __restrict__ out) {
  __shared__ union {
    struct { __bf16 A[128 * 72]; __bf16 B[128 * 72]; } ab;  // 36864 B
    float Cs[64 * 132];                                      // 33792 B
  } sm;
  const int tid = threadIdx.x, lane = tid & 31, wv = tid >> 5;
  const int wm = wv >> 1, wn = wv & 1;
  const int b = blockIdx.z, m0 = blockIdx.y * 128, n0 = blockIdx.x * 128;
  v8f acc[2][4] = {};
  const __bf16* Orow = Ob + ((size_t)b * T_ + m0) * C_;
  const __bf16* wrow = woutT + (size_t)n0 * C_;
#pragma unroll 1
  for (int kt = 0; kt < C_ / 64; ++kt) {
    const __bf16* osrc = Orow + kt * 64;
    const __bf16* wsrc = wrow + kt * 64;
#pragma unroll
    for (int i = 0; i < 4; ++i) {
      int idx = tid + i * 256;
      int m = idx >> 3, kv = (idx & 7) * 8;
      *(uint4*)(sm.ab.A + m * 72 + kv) = *(const uint4*)(osrc + (size_t)m * C_ + kv);
      *(uint4*)(sm.ab.B + m * 72 + kv) = *(const uint4*)(wsrc + (size_t)m * C_ + kv);
    }
    __syncthreads();
#pragma unroll
    for (int kc = 0; kc < 2; ++kc) {
      v16bf a0 = load_a_frag(sm.ab.A + (wm * 32 + 0) * 72 + kc * 32, 72, lane);
      v16bf a1 = load_a_frag(sm.ab.A + (wm * 32 + 16) * 72 + kc * 32, 72, lane);
#pragma unroll
      for (int j = 0; j < 4; ++j) {
        v16bf bf = load_b_frag(sm.ab.B + (wn * 64 + j * 16) * 72 + kc * 32, 72, lane);
        acc[0][j] = wmma_bf16(a0, bf, acc[0][j]);
        acc[1][j] = wmma_bf16(a1, bf, acc[1][j]);
      }
    }
    __syncthreads();
  }
  const int hl = lane >> 4, nn = lane & 15;
#pragma unroll 1
  for (int pass = 0; pass < 2; ++pass) {
    if (wn == pass) {
#pragma unroll
      for (int j = 0; j < 4; ++j) {
        int nl = j * 16 + nn;           // 0..63 within this pass
        float bv = bias[n0 + pass * 64 + nl];
#pragma unroll
        for (int i = 0; i < 2; ++i)
#pragma unroll
          for (int r = 0; r < 8; ++r)
            sm.Cs[nl * 132 + wm * 32 + i * 16 + r + hl * 8] = acc[i][j][r] + bv;
      }
    }
    __syncthreads();
#pragma unroll
    for (int i = 0; i < 8; ++i) {       // 64 rows(c) x 32 float4(t)
      int idx = tid + i * 256;
      int nl = idx >> 5, mv = (idx & 31) * 4;
      float4 f = *(const float4*)(sm.Cs + nl * 132 + mv);
      *(float4*)(out + ((size_t)b * C_ + n0 + pass * 64 + nl) * T_ + m0 + mv) = f;
    }
    __syncthreads();
  }
}

// ---------------------------------------------------------------------------
extern "C" void kernel_launch(void* const* d_in, const int* in_sizes, int n_in,
                              void* d_out, int out_size, void* d_ws, size_t ws_size,
                              hipStream_t stream) {
  (void)in_sizes; (void)n_in; (void)out_size; (void)ws_size;
  const float* x     = (const float*)d_in[0];
  const float* w_qkv = (const float*)d_in[1];
  const float* b_qkv = (const float*)d_in[2];
  const float* w_out = (const float*)d_in[3];
  const float* b_out = (const float*)d_in[4];
  float* out = (float*)d_out;

  const size_t nTC = (size_t)T_ * C_;          // 2,097,152
  __bf16* xbf   = (__bf16*)d_ws;               // [B][T][C]
  __bf16* wqkvT = xbf + (size_t)B_ * nTC;      // [3C][C]
  __bf16* woutT = wqkvT + (size_t)N3_ * C_;    // [C][C]
  __bf16* Qb    = woutT + (size_t)C_ * C_;     // [B][H][T][D]
  __bf16* Kb    = Qb + (size_t)B_ * nTC;
  __bf16* Vb    = Kb + (size_t)B_ * nTC;
  __bf16* Ob    = Vb + (size_t)B_ * nTC;       // [B][T][C]

  // Pre-pass: transpose + convert to bf16.
  tcvt_kernel<<<dim3(T_ / 32, C_ / 32, B_), 256, 0, stream>>>(
      x, xbf, C_, T_, (size_t)C_ * T_, nTC);
  tcvt_kernel<<<dim3(N3_ / 32, C_ / 32, 1), 256, 0, stream>>>(
      w_qkv, wqkvT, C_, N3_, 0, 0);
  tcvt_kernel<<<dim3(C_ / 32, C_ / 32, 1), 256, 0, stream>>>(
      w_out, woutT, C_, C_, 0, 0);

  qkv_gemm_kernel<<<dim3(N3_ / 128, T_ / 128, B_), 256, 0, stream>>>(
      xbf, wqkvT, b_qkv, Qb, Kb, Vb);
  attn_kernel<<<dim3(T_ / 128, H_, B_), 256, 0, stream>>>(Qb, Kb, Vb, Ob);
  oproj_kernel<<<dim3(C_ / 128, T_ / 128, B_), 256, 0, stream>>>(
      Ob, woutT, b_out, out);
}
